// GATRecommender_78331613545097
// MI455X (gfx1250) — compile-verified, matrix operand order checked by reading, not run on hardware
//
#include <hip/hip_runtime.h>
#include <hip/hip_bf16.h>
#include <math.h>

// ---------------------------------------------------------------------------
// GATv2 recommender (3 layers) for gfx1250 / MI455X.
//  - GEMMs: v_wmma_f32_16x16x32_bf16, one wave per 16x16 tile, K=128 fully
//    unrolled -> 4 chained WMMAs, vector (b128) fragment loads.
//  - Edge phase: L2-resident gather/atomic streaming (xl/xr = 25.6MB << 192MB L2),
//    templated on (H,C) to remove integer divides.
// ---------------------------------------------------------------------------

typedef __attribute__((ext_vector_type(16))) __bf16 v16bf;
typedef __attribute__((ext_vector_type(8)))  __bf16 bf16x8;
typedef __attribute__((ext_vector_type(8)))  float  v8f;

#define NEG_SLOPE 0.2f
#define LN_EPS    1e-5f

// ---------------- utility kernels ----------------

__global__ void fill_f32(float* __restrict__ p, float v, int n) {
  int i = blockIdx.x * blockDim.x + threadIdx.x;
  if (i < n) p[i] = v;
}

// W fp32 [K, Nd] row-major  ->  Wt bf16 [Nd, K] row-major (K contiguous)
__global__ void convert_transpose_w(const float* __restrict__ W,
                                    __bf16* __restrict__ Wt, int K, int Nd) {
  int i = blockIdx.x * blockDim.x + threadIdx.x;
  if (i >= K * Nd) return;
  int k = i / Nd, n = i - k * Nd;
  Wt[(size_t)n * K + k] = (__bf16)W[i];
}

// ---------------- WMMA GEMM:  Y[M,Nd] = X[M,K] @ W[K,Nd] + bias ----------------
// Wt is bf16 pre-transposed weight [Nd, K]. One wave per 16x16 tile; M % 16 == 0.
// A-frag (16x32 bf16): lane<16 holds row M=lane, K {kb+0..7, kb+16..23};
//                      lane>=16 holds row M=lane-16, K {kb+8..15, kb+24..31}.
// B-frag mirrors with lane = output column. C/D: VGPR v -> row half*8+v, col=lane&15.
template <int KC, int NDC>
__global__ void gemm_wmma_bf16(const float* __restrict__ X,
                               const __bf16* __restrict__ Wt,
                               const float* __restrict__ bias,
                               float* __restrict__ Y, int M) {
  const int tiles_n = NDC >> 4;
  int wave = (blockIdx.x * blockDim.x + threadIdx.x) >> 5;
  int lane = threadIdx.x & 31;
  int tiles_m = M >> 4;
  if (wave >= tiles_m * tiles_n) return;           // wave-uniform exit (EXEC stays all-1)
  int tm = wave / tiles_n;
  int tn = wave - tm * tiles_n;

  int half = lane >> 4;
  int r    = lane & 15;
  int row  = tm * 16 + r;
  int col  = tn * 16 + r;

  // A: four 16B float4 loads (two 8-float runs) per K-block; B: two 16B bf16x8 groups per K-block.
  const float4*  xq = (const float4*)(X + (size_t)row * KC + half * 8);
  const bf16x8*  bq = (const bf16x8*)(Wt + (size_t)col * KC + half * 8);

  v8f acc = {};
#pragma unroll
  for (int kb = 0; kb < KC / 32; ++kb) {
    float4 f0 = xq[kb * 8 + 0];
    float4 f1 = xq[kb * 8 + 1];
    float4 f2 = xq[kb * 8 + 4];   // +16 floats
    float4 f3 = xq[kb * 8 + 5];
    bf16x8 b0 = bq[kb * 4 + 0];
    bf16x8 b1 = bq[kb * 4 + 2];   // +16 bf16

    v16bf a;
    a[0] = (__bf16)f0.x; a[1] = (__bf16)f0.y; a[2]  = (__bf16)f0.z; a[3]  = (__bf16)f0.w;
    a[4] = (__bf16)f1.x; a[5] = (__bf16)f1.y; a[6]  = (__bf16)f1.z; a[7]  = (__bf16)f1.w;
    a[8] = (__bf16)f2.x; a[9] = (__bf16)f2.y; a[10] = (__bf16)f2.z; a[11] = (__bf16)f2.w;
    a[12]= (__bf16)f3.x; a[13]= (__bf16)f3.y; a[14] = (__bf16)f3.z; a[15] = (__bf16)f3.w;
    v16bf b = __builtin_shufflevector(b0, b1, 0, 1, 2, 3, 4, 5, 6, 7,
                                      8, 9, 10, 11, 12, 13, 14, 15);

    acc = __builtin_amdgcn_wmma_f32_16x16x32_bf16(
        /*neg_a=*/false, a, /*neg_b=*/false, b,
        /*c_mod=*/(short)0, acc, /*reuse_a=*/false, /*reuse_b=*/false);
  }

  float bv = bias[col];
  float* yp = Y + (size_t)(tm * 16 + half * 8) * NDC + col;
#pragma unroll
  for (int v = 0; v < 8; ++v) {
    *yp = acc[v] + bv;
    yp += NDC;
  }
}

// ---------------- edge phase ----------------

__device__ __forceinline__ void atomicMaxFloat(float* addr, float val) {
  unsigned int* ua = (unsigned int*)addr;
  unsigned int old = __float_as_uint(*addr);
  while (__uint_as_float(old) < val) {
    unsigned int assumed = old;
    old = atomicCAS(ua, assumed, __float_as_uint(val));
    if (old == assumed) break;
  }
}

// one thread per edge: score[e,h] = sum_c lrelu(xl[src]+xr[dst]+ea*We)*att ; smax = segmax
template <int H, int C>
__global__ void score_kernel(const float* __restrict__ xl, const float* __restrict__ xr,
                             const int* __restrict__ src, const int* __restrict__ dst,
                             const float* __restrict__ eattr,
                             const float* __restrict__ We, const float* __restrict__ att,
                             float* __restrict__ score, float* __restrict__ smax, int E) {
  constexpr int D = H * C;
  int e = blockIdx.x * blockDim.x + threadIdx.x;
  if (e >= E) return;
  int s = src[e], t = dst[e];
  const float4* pl = (const float4*)(xl + (size_t)s * D);
  const float4* pr = (const float4*)(xr + (size_t)t * D);
  __builtin_prefetch(pl, 0, 3);                    // global_prefetch_b8
  __builtin_prefetch(pr, 0, 3);
  float ea = eattr[e];
#pragma unroll
  for (int h = 0; h < H; ++h) {
    float acc = 0.f;
#pragma unroll 2
    for (int q = 0; q < C / 4; ++q) {
      int g = h * (C / 4) + q;
      float4 l4 = pl[g], r4 = pr[g];
      const float4 w4 = ((const float4*)We)[g];
      const float4 a4 = ((const float4*)att)[g];
      float v;
      v = l4.x + r4.x + ea * w4.x; v = v > 0.f ? v : NEG_SLOPE * v; acc += v * a4.x;
      v = l4.y + r4.y + ea * w4.y; v = v > 0.f ? v : NEG_SLOPE * v; acc += v * a4.y;
      v = l4.z + r4.z + ea * w4.z; v = v > 0.f ? v : NEG_SLOPE * v; acc += v * a4.z;
      v = l4.w + r4.w + ea * w4.w; v = v > 0.f ? v : NEG_SLOPE * v; acc += v * a4.w;
    }
    score[(size_t)e * H + h] = acc;
    atomicMaxFloat(&smax[(size_t)t * H + h], acc);
  }
}

// one thread per (edge,head): a = exp(score - smax[dst]); denom[dst] += a (in-place over score)
template <int H>
__global__ void alpha_kernel(float* __restrict__ score, const float* __restrict__ smax,
                             float* __restrict__ denom, const int* __restrict__ dst, int E) {
  int i = blockIdx.x * blockDim.x + threadIdx.x;
  if (i >= E * H) return;
  int e = i / H, h = i - e * H;
  int t = dst[e];
  float a = __expf(score[i] - smax[t * H + h]);
  score[i] = a;
  atomicAdd(&denom[t * H + h], a);
}

// one thread per (edge, feature): agg[dst, j] += a[e,h(j)] * xl[src, j]
template <int H, int C>
__global__ void agg_kernel(const float* __restrict__ a, const float* __restrict__ xl,
                           const int* __restrict__ src, const int* __restrict__ dst,
                           float* __restrict__ agg, int E) {
  constexpr int D = H * C;
  int i = blockIdx.x * blockDim.x + threadIdx.x;
  if (i >= E * D) return;
  int e = i / D;          // D is a power of two -> shift
  int j = i - e * D;
  int h = j / C;          // C = 32 -> shift
  int s = src[e], t = dst[e];
  atomicAdd(&agg[(size_t)t * D + j], a[e * H + h] * xl[(size_t)s * D + j]);
}

// one wave32 per node: normalize / (concat|mean) / +bias / ELU / LayerNorm / +residual
__global__ void finalize_kernel(const float* __restrict__ agg, const float* __restrict__ denom,
                                const float* __restrict__ bias, const float* __restrict__ gamma,
                                const float* __restrict__ beta, const float* __restrict__ xres,
                                float* __restrict__ xout,
                                int Nn, int H, int C, int concat) {
  int wv   = (blockIdx.x * blockDim.x + threadIdx.x) >> 5;
  int lane = threadIdx.x & 31;
  if (wv >= Nn) return;                            // wave-uniform
  int d    = H * C;
  int dout = concat ? d : C;
  int per  = dout >> 5;                            // 4 (dout=128) or 1 (dout=32)
  const float* ag = agg   + (size_t)wv * d;
  const float* dn = denom + (size_t)wv * H;
  float v[4];
  for (int p = 0; p < per; ++p) {
    int j = p * 32 + lane;
    float o;
    if (concat) {
      o = ag[j] / (dn[j / C] + 1e-16f);
    } else {
      float sm = 0.f;
      for (int h = 0; h < H; ++h) sm += ag[h * C + j] / (dn[h] + 1e-16f);
      o = sm / (float)H;
    }
    o += bias[j];
    v[p] = o > 0.f ? o : (__expf(o) - 1.f);        // ELU
  }
  float sum = 0.f;
  for (int p = 0; p < per; ++p) sum += v[p];
  for (int o = 16; o > 0; o >>= 1) sum += __shfl_xor(sum, o, 32);
  float mu = sum / (float)dout;
  float var = 0.f;
  for (int p = 0; p < per; ++p) { float df = v[p] - mu; var += df * df; }
  for (int o = 16; o > 0; o >>= 1) var += __shfl_xor(var, o, 32);
  var /= (float)dout;
  float inv = rsqrtf(var + LN_EPS);
  for (int p = 0; p < per; ++p) {
    int j = p * 32 + lane;
    xout[(size_t)wv * dout + j] =
        (v[p] - mu) * inv * gamma[j] + beta[j] + xres[(size_t)wv * dout + j];
  }
}

// ---------------- host orchestration ----------------

static inline int cdiv(int a, int b) { return (a + b - 1) / b; }

extern "C" void kernel_launch(void* const* d_in, const int* in_sizes, int n_in,
                              void* d_out, int out_size, void* d_ws, size_t ws_size,
                              hipStream_t stream) {
  (void)n_in; (void)out_size; (void)ws_size;
  const int IN_DIM = 128;
  const float* node_emb = (const float*)d_in[0];
  const int Nn = in_sizes[0] / IN_DIM;

  // params flattened in setup_inputs() insertion order
  const float *Wl[3], *bl[3], *Wr[3], *br[3], *We[3], *att[3], *bias[3], *gamma[3], *beta[3];
  const float *projW = nullptr, *projb = nullptr;
  int idx = 1;
  for (int L = 0; L < 3; ++L) {
    Wl[L]   = (const float*)d_in[idx++];  bl[L]   = (const float*)d_in[idx++];
    Wr[L]   = (const float*)d_in[idx++];  br[L]   = (const float*)d_in[idx++];
    We[L]   = (const float*)d_in[idx++];  att[L]  = (const float*)d_in[idx++];
    bias[L] = (const float*)d_in[idx++];
    gamma[L]= (const float*)d_in[idx++];  beta[L] = (const float*)d_in[idx++];
    if (L == 2) { projW = (const float*)d_in[idx++]; projb = (const float*)d_in[idx++]; }
  }
  const int* ei = (const int*)d_in[idx];
  const int  E  = in_sizes[idx] / 2;  idx++;
  const float* eattr = (const float*)d_in[idx++];
  const int* srcp = ei;
  const int* dstp = ei + E;

  // workspace carve-up (~120 MB)
  size_t off = 0;
  auto wsalloc = [&](size_t bytes) -> char* {
    char* p = (char*)d_ws + off;
    off += (bytes + 255) & ~(size_t)255;
    return p;
  };
  float* xbuf0 = (float*)wsalloc((size_t)Nn * 128 * 4);
  float* xbuf1 = (float*)wsalloc((size_t)Nn * 128 * 4);
  float* xl    = (float*)wsalloc((size_t)Nn * 128 * 4);
  float* xr    = (float*)wsalloc((size_t)Nn * 128 * 4);   // reused as agg buffer
  float* xres  = (float*)wsalloc((size_t)Nn * 32 * 4);
  float* sc    = (float*)wsalloc((size_t)E * 4 * 4);
  float* smax  = (float*)wsalloc((size_t)Nn * 4 * 4);
  float* denom = (float*)wsalloc((size_t)Nn * 4 * 4);
  __bf16* WlT[3]; __bf16* WrT[3];
  for (int L = 0; L < 3; ++L) {
    int dd = (L == 2) ? 32 : 128;
    WlT[L] = (__bf16*)wsalloc((size_t)128 * dd * 2);
    WrT[L] = (__bf16*)wsalloc((size_t)128 * dd * 2);
  }
  __bf16* WpT = (__bf16*)wsalloc((size_t)128 * 32 * 2);

  // weight convert + transpose (tiny)
  for (int L = 0; L < 3; ++L) {
    int dd = (L == 2) ? 32 : 128;
    int n = 128 * dd;
    convert_transpose_w<<<cdiv(n, 256), 256, 0, stream>>>(Wl[L], WlT[L], 128, dd);
    convert_transpose_w<<<cdiv(n, 256), 256, 0, stream>>>(Wr[L], WrT[L], 128, dd);
  }
  convert_transpose_w<<<cdiv(128 * 32, 256), 256, 0, stream>>>(projW, WpT, 128, 32);

  const float* xin = node_emb;
  for (int L = 0; L < 3; ++L) {
    const int H = (L == 2) ? 1 : 4;
    const int C = 32;
    const int d = H * C;            // 128,128,32
    const int concat = (L != 2);

    // xl = x@Wl+bl, xr = x@Wr+br  (WMMA)
    int waves = (Nn / 16) * (d >> 4);
    int gblocks = cdiv(waves * 32, 256);
    if (L != 2) {
      gemm_wmma_bf16<128, 128><<<gblocks, 256, 0, stream>>>(xin, WlT[L], bl[L], xl, Nn);
      gemm_wmma_bf16<128, 128><<<gblocks, 256, 0, stream>>>(xin, WrT[L], br[L], xr, Nn);
    } else {
      gemm_wmma_bf16<128, 32><<<gblocks, 256, 0, stream>>>(xin, WlT[L], bl[L], xl, Nn);
      gemm_wmma_bf16<128, 32><<<gblocks, 256, 0, stream>>>(xin, WrT[L], br[L], xr, Nn);
      gemm_wmma_bf16<128, 32><<<gblocks, 256, 0, stream>>>(xin, WpT, projb, xres, Nn);
    }

    fill_f32<<<cdiv(Nn * H, 256), 256, 0, stream>>>(smax, -3.4e38f, Nn * H);
    fill_f32<<<cdiv(Nn * H, 256), 256, 0, stream>>>(denom, 0.f, Nn * H);

    if (L != 2) {
      score_kernel<4, 32><<<cdiv(E, 256), 256, 0, stream>>>(xl, xr, srcp, dstp, eattr,
                                                            We[L], att[L], sc, smax, E);
      alpha_kernel<4><<<cdiv(E * 4, 256), 256, 0, stream>>>(sc, smax, denom, dstp, E);
    } else {
      score_kernel<1, 32><<<cdiv(E, 256), 256, 0, stream>>>(xl, xr, srcp, dstp, eattr,
                                                            We[L], att[L], sc, smax, E);
      alpha_kernel<1><<<cdiv(E, 256), 256, 0, stream>>>(sc, smax, denom, dstp, E);
    }

    float* agg = xr;                // xr no longer needed -> reuse as aggregation buffer
    fill_f32<<<cdiv(Nn * d, 256), 256, 0, stream>>>(agg, 0.f, Nn * d);
    if (L != 2) {
      agg_kernel<4, 32><<<cdiv(E * 128, 256), 256, 0, stream>>>(sc, xl, srcp, dstp, agg, E);
    } else {
      agg_kernel<1, 32><<<cdiv(E * 32, 256), 256, 0, stream>>>(sc, xl, srcp, dstp, agg, E);
    }

    float* xout = (L == 2) ? (float*)d_out : (L == 0 ? xbuf0 : xbuf1);
    const float* res = (L == 2) ? xres : xin;    // din==dout==128 for layers 0/1
    finalize_kernel<<<cdiv(Nn * 32, 256), 256, 0, stream>>>(agg, denom, bias[L], gamma[L],
                                                            beta[L], res, xout, Nn, H, C, concat);
    xin = xout;
  }
}
